// Alignment_49735721287759
// MI455X (gfx1250) — matrix operand order repaired and checked by reading, not compiled, not run on hardware
//
#include <hip/hip_runtime.h>
#include <cstdint>
#include <cstddef>

// ---------------------------------------------------------------------------
// CLIP-style token-selection cross-attention alignment for gfx1250 (MI455X).
// Heavy GEMMs (QKV / out-proj / MLP) run on v_wmma_f32_16x16x32_f16.
// ---------------------------------------------------------------------------

typedef _Float16 half_t;
typedef __attribute__((ext_vector_type(16))) _Float16 v16h;
typedef __attribute__((ext_vector_type(8)))  _Float16 v8h;
typedef __attribute__((ext_vector_type(8)))  float    v8f;

#define BN      32      // batch
#define PATCH   196
#define SEQ     77
#define DIM     512
#define NLAYERS 2
#define TOPK    16
#define NHEADS  8
#define HEADD   64
#define LQ      17      // cls + 16 selected tokens
#define XS      520     // padded f16 row stride (halfs): (520/2)%64==4 -> conflict-free b128 rows

#define IMG_ROWS (BN*PATCH)   // 6272
#define TXT_ROWS (BN*SEQ)     // 2464

// LDS layout (dynamic) for the cross-attention kernel
#define QLD_BYTES (LQ*DIM*4)                 // 34816  f32 residual q
#define XA_BYTES  (32*XS*2)                  // 33280  padded f16 A operand
#define KQV_BYTES (3*LQ*XS*2)                // 53040  K,V,Q f16
#define SC_BYTES  (NHEADS*LQ*LQ*4)           // 9248   attention probs f32
#define R2_BYTES  (KQV_BYTES + SC_BYTES)     // 62288  (xB 33280 overlays this region)
#define SMEM_BYTES (QLD_BYTES + XA_BYTES + R2_BYTES)   // 130384

// ---------------------------------------------------------------------------
__device__ __forceinline__ float waveReduceSum(float x) {
#pragma unroll
  for (int off = 16; off > 0; off >>= 1) x += __shfl_xor(x, off, 32);
  return x;
}

// LayerNorm one 512-wide row (one wave), write f16 into padded A buffer row.
__device__ __forceinline__ void ln_row_to_f16(const float* __restrict__ src,
                                              const float* __restrict__ w,
                                              const float* __restrict__ b,
                                              half_t* __restrict__ dst, int lane) {
  float v[16];
  float s = 0.f, s2 = 0.f;
#pragma unroll
  for (int t = 0; t < 16; ++t) {
    float x = src[lane * 16 + t];
    v[t] = x; s += x; s2 += x * x;
  }
  s  = waveReduceSum(s);
  s2 = waveReduceSum(s2);
  float mean = s * (1.f / 512.f);
  float var  = s2 * (1.f / 512.f) - mean * mean;
  float rstd = rsqrtf(var + 1e-5f);
#pragma unroll
  for (int t = 0; t < 16; ++t) {
    int c = lane * 16 + t;
    dst[c] = (half_t)(((v[t] - mean) * rstd) * w[c] + b[c]);
  }
}

// A fragment: 16x32 f16 (M x K), per ISA layout.
// lane L: M = L%16 ; K chunks {kb..kb+7, kb+16..kb+23} with kb = (L<16?0:8)
__device__ __forceinline__ v16h load_a(const half_t* __restrict__ A, int mt, int k0, int lane) {
  const half_t* p = A + (size_t)(mt * 16 + (lane & 15)) * XS + k0 + ((lane >> 4) << 3);
  v8h lo = *(const v8h*)p;
  v8h hi = *(const v8h*)(p + 16);
  v16h a;
#pragma unroll
  for (int t = 0; t < 8; ++t) { a[t] = lo[t]; a[t + 8] = hi[t]; }
  return a;
}

// C(32 x 512) += A(32 x Kdim) * W^T ; W is f16 row-major [Nout][ldw] (opt. K-slice kcol0).
// 8 waves, each wave owns 8 tiles: n-tiles wave*4..wave*4+3 for both M-tiles.
// All 4 B fragments of a k-step are loaded up front so the compiler can emit
// one load clause and drain it incrementally across the 8 WMMAs (latency hiding).
__device__ __forceinline__ void gemm_acc(const half_t* __restrict__ Alds,
                                         const half_t* __restrict__ W,
                                         int ldw, int kcol0, int Kdim,
                                         v8f* acc, int wave, int lane) {
  const int nrow0 = (wave * 4) * 16 + (lane & 15);
  const int ksel  = ((lane >> 4) << 3);
  const half_t* Wbase = W + (size_t)nrow0 * ldw + kcol0 + ksel;
  for (int k0 = 0; k0 < Kdim; k0 += 32) {
    v16h bf[4];
#pragma unroll
    for (int c = 0; c < 4; ++c) {
      // B[k][n] = W[n][kcol0+k]: lane reads k chunks {kb..kb+7, kb+16..kb+23} (16B aligned).
      const half_t* wp = Wbase + (size_t)c * 16 * ldw + k0;
      v8h blo = *(const v8h*)wp;
      v8h bhi = *(const v8h*)(wp + 16);
#pragma unroll
      for (int t = 0; t < 8; ++t) { bf[c][t] = blo[t]; bf[c][t + 8] = bhi[t]; }
    }
    v16h a0 = load_a(Alds, 0, k0, lane);
    v16h a1 = load_a(Alds, 1, k0, lane);
#pragma unroll
    for (int c = 0; c < 4; ++c) {
      acc[c]     = __builtin_amdgcn_wmma_f32_16x16x32_f16(false, a0, false, bf[c], (short)0, acc[c],     false, false);
      acc[4 + c] = __builtin_amdgcn_wmma_f32_16x16x32_f16(false, a1, false, bf[c], (short)0, acc[4 + c], false, false);
    }
  }
}

// Store valid rows (M<17) of the 32x512 result as f16 (+bias, optional SiLU(1.702)).
__device__ __forceinline__ void store_tiles_f16(half_t* __restrict__ dst, const v8f* acc,
                                                const float* __restrict__ bias, bool silu,
                                                int wave, int lane) {
  int nlo  = lane & 15;
  int mofs = (lane >> 4) << 3;
#pragma unroll
  for (int c = 0; c < 8; ++c) {
    int mt = c >> 2;
    int n  = (wave * 4 + (c & 3)) * 16 + nlo;
    float bv = bias[n];
#pragma unroll
    for (int r = 0; r < 8; ++r) {
      int M = mt * 16 + r + mofs;
      if (M < LQ) {
        float x = acc[c][r] + bv;
        if (silu) x = x * (1.f / (1.f + expf(-1.702f * x)));
        dst[(size_t)M * XS + n] = (half_t)x;
      }
    }
  }
}

// q[M][n] += acc + bias for valid rows.
__device__ __forceinline__ void add_tiles_to_q(float* __restrict__ qld, const v8f* acc,
                                               const float* __restrict__ bias,
                                               int wave, int lane) {
  int nlo  = lane & 15;
  int mofs = (lane >> 4) << 3;
#pragma unroll
  for (int c = 0; c < 8; ++c) {
    int mt = c >> 2;
    int n  = (wave * 4 + (c & 3)) * 16 + nlo;
    float bv = bias[n];
#pragma unroll
    for (int r = 0; r < 8; ++r) {
      int M = mt * 16 + r + mofs;
      if (M < LQ) qld[(size_t)M * DIM + n] += acc[c][r] + bv;
    }
  }
}

// ---------------------------------------------------------------------------
// Prep kernels
// ---------------------------------------------------------------------------
__global__ __launch_bounds__(256) void al_f32_to_f16(const float* __restrict__ src,
                                                     half_t* __restrict__ dst, int n) {
  for (int e = blockIdx.x * 256 + threadIdx.x; e < n; e += gridDim.x * 256)
    dst[e] = (half_t)src[e];
}

// L2-normalize every token row (one wave per 512-wide row).
__global__ __launch_bounds__(256) void al_normalize(const float* __restrict__ img_t,
                                                    const float* __restrict__ txt_t,
                                                    float* __restrict__ img_n,
                                                    float* __restrict__ txt_n) {
  int gw   = (blockIdx.x * 256 + threadIdx.x) >> 5;
  int lane = threadIdx.x & 31;
  if (gw >= IMG_ROWS + TXT_ROWS) return;
  const float* src; float* dst;
  if (gw < IMG_ROWS) { src = img_t + (size_t)gw * DIM; dst = img_n + (size_t)gw * DIM; }
  else { int r = gw - IMG_ROWS; src = txt_t + (size_t)r * DIM; dst = txt_n + (size_t)r * DIM; }
  float s2 = 0.f;
#pragma unroll
  for (int t = 0; t < 16; ++t) { float x = src[lane + 32 * t]; s2 += x * x; }
  s2 = waveReduceSum(s2);
  float rn = rsqrtf(s2);
#pragma unroll
  for (int t = 0; t < 16; ++t) { int c = lane + 32 * t; dst[c] = src[c] * rn; }
}

// t_masked[j] = sum_s mask[j,s]*txt_n[j,s]; img_sum[i] = sum_p img_n[i,p]
__global__ __launch_bounds__(256) void al_masksum(const float* __restrict__ txt_n,
                                                  const float* __restrict__ img_n,
                                                  const int* __restrict__ mask,
                                                  float* __restrict__ t_masked,
                                                  float* __restrict__ img_sum) {
  int b = blockIdx.x, tid = threadIdx.x;
  if (b < BN) {
    for (int c = tid; c < DIM; c += 256) {
      float s = 0.f;
      for (int t = 0; t < SEQ; ++t)
        s += (float)mask[b * SEQ + t] * txt_n[((size_t)b * SEQ + t) * DIM + c];
      t_masked[(size_t)b * DIM + c] = s;
    }
  } else {
    int i = b - BN;
    for (int c = tid; c < DIM; c += 256) {
      float s = 0.f;
      for (int p = 0; p < PATCH; ++p)
        s += img_n[((size_t)i * PATCH + p) * DIM + c];
      img_sum[(size_t)i * DIM + c] = s;
    }
  }
}

// Score dots (sim never materialized). One wave per output element.
__global__ __launch_bounds__(256) void al_scores(const float* __restrict__ img_n,
                                                 const float* __restrict__ txt_n,
                                                 const float* __restrict__ t_masked,
                                                 const float* __restrict__ img_sum,
                                                 float* __restrict__ img_sc,
                                                 float* __restrict__ txt_sc) {
  int gw   = (blockIdx.x * 256 + threadIdx.x) >> 5;
  int lane = threadIdx.x & 31;
  const float *a, *b; float* o; int oi;
  const int NIMG = BN * BN * PATCH;        // 200704
  const int NTXT = BN * BN * SEQ;          // 78848
  if (gw < NIMG) {
    int ij = gw / PATCH; int p = gw - ij * PATCH;
    int i = ij >> 5; int j = ij & 31;
    a = img_n + ((size_t)i * PATCH + p) * DIM;
    b = t_masked + (size_t)j * DIM;
    o = img_sc; oi = gw;
  } else if (gw < NIMG + NTXT) {
    int g = gw - NIMG;
    int ij = g / SEQ; int s = g - ij * SEQ;
    int i = ij >> 5; int j = ij & 31;
    a = img_sum + (size_t)i * DIM;
    b = txt_n + ((size_t)j * SEQ + s) * DIM;
    o = txt_sc; oi = g;
  } else return;
  float sum = 0.f;
#pragma unroll
  for (int t = 0; t < 16; ++t) { int c = lane + 32 * t; sum += a[c] * b[c]; }
  sum = waveReduceSum(sum);
  if (lane == 0) o[oi] = sum;
}

// top-16 by value (ties -> lowest index), then sort indices ascending.
__device__ void al_topk_sel(const float* __restrict__ v, int n, int* __restrict__ out) {
  unsigned taken[7] = {0, 0, 0, 0, 0, 0, 0};
  int idxs[TOPK];
  for (int kk = 0; kk < TOPK; ++kk) {
    float best = 0.f; int bi = 0; bool found = false;
    for (int e = 0; e < n; ++e) {
      if ((taken[e >> 5] >> (e & 31)) & 1u) continue;
      float x = v[e];
      if (!found || x > best) { best = x; bi = e; found = true; }
    }
    taken[bi >> 5] |= 1u << (bi & 31);
    idxs[kk] = bi;
  }
  for (int a = 1; a < TOPK; ++a) {
    int key = idxs[a], b = a - 1;
    while (b >= 0 && idxs[b] > key) { idxs[b + 1] = idxs[b]; --b; }
    idxs[b + 1] = key;
  }
  for (int kk = 0; kk < TOPK; ++kk) out[kk] = idxs[kk];
}

__global__ __launch_bounds__(256) void al_topk(const float* __restrict__ img_sc,
                                               const float* __restrict__ txt_sc,
                                               int* __restrict__ idx_i,
                                               int* __restrict__ idx_t) {
  int t = blockIdx.x * 256 + threadIdx.x;
  if (t < BN * BN) al_topk_sel(img_sc + (size_t)t * PATCH, PATCH, idx_i + t * TOPK);
  else if (t < 2 * BN * BN) {
    int u = t - BN * BN;
    al_topk_sel(txt_sc + (size_t)u * SEQ, SEQ, idx_t + u * TOPK);
  }
}

// ---------------------------------------------------------------------------
// Cross-attention mega-kernel: 1 block = one (pair, direction); 2 layers.
// ---------------------------------------------------------------------------
__global__ __launch_bounds__(256) void al_cross_attn(
    const float* __restrict__ image_feature, const float* __restrict__ text_feature,
    const float* __restrict__ img_cls, const float* __restrict__ txt_cls,
    const float* __restrict__ in_proj_b, const float* __restrict__ out_b,
    const float* __restrict__ ln1_w, const float* __restrict__ ln1_b,
    const float* __restrict__ ln2_w, const float* __restrict__ ln2_b,
    const float* __restrict__ ln3_w, const float* __restrict__ ln3_b,
    const float* __restrict__ fc_b, const float* __restrict__ proj_b,
    const float* __restrict__ img_n, const float* __restrict__ txt_n,
    const int* __restrict__ idx_i, const int* __restrict__ idx_t,
    const half_t* __restrict__ Wh_in, const half_t* __restrict__ Wh_out,
    const half_t* __restrict__ Wh_fc, const half_t* __restrict__ Wh_pj,
    float* __restrict__ out) {
  extern __shared__ char smem[];
  float*  qld = (float*)smem;                       // [17][512] f32 residual
  half_t* xA  = (half_t*)(smem + QLD_BYTES);        // [32][520] f16 A operand
  char*   r2  = smem + QLD_BYTES + XA_BYTES;
  half_t* Kf  = (half_t*)r2;                        // [17][520] f16
  half_t* Vf  = Kf + LQ * XS;
  half_t* Qf  = Vf + LQ * XS;
  float*  sc  = (float*)(r2 + KQV_BYTES);           // [8][17][17] f32
  half_t* xB  = (half_t*)r2;                        // [32][520] f16 (overlays K/V/Q)

  const int tid  = threadIdx.x;
  const int wave = tid >> 5;
  const int lane = tid & 31;
  const int bid  = blockIdx.x;
  const int dir  = bid >> 10;          // 0 = final_img, 1 = final_txt
  const int pr   = bid & 1023;
  const int i    = pr >> 5;
  const int j    = pr & 31;

  const int* idxI = idx_i + pr * TOPK;
  const int* idxT = idx_t + pr * TOPK;

  auto qrow = [&](int r) -> const float* {
    if (dir == 0)
      return (r == 0) ? img_cls : img_n + ((size_t)(i * PATCH + idxI[r - 1])) * DIM;
    else
      return (r == 0) ? txt_cls : txt_n + ((size_t)(j * SEQ + idxT[r - 1])) * DIM;
  };
  auto krow = [&](int r) -> const float* {
    if (dir == 0)
      return (r == 0) ? text_feature + (size_t)j * DIM
                      : txt_n + ((size_t)(j * SEQ + idxT[r - 1])) * DIM;
    else
      return (r == 0) ? image_feature + (size_t)i * DIM
                      : img_n + ((size_t)(i * PATCH + idxI[r - 1])) * DIM;
  };

  // zero pad rows 17..31 of xA (never rewritten afterwards)
  for (int e = tid; e < (32 - LQ) * XS; e += 256) xA[LQ * XS + e] = (half_t)0.f;
  // gather q0 into LDS f32
  for (int r = wave; r < LQ; r += 8) {
    const float* src = qrow(r);
#pragma unroll
    for (int t = 0; t < 16; ++t) qld[(size_t)r * DIM + lane * 16 + t] = src[lane * 16 + t];
  }
  __syncthreads();

  for (int l = 0; l < NLAYERS; ++l) {
    const half_t* Wq  = Wh_in + (size_t)l * 3 * DIM * DIM;
    const half_t* Wk  = Wq + (size_t)DIM * DIM;
    const half_t* Wv  = Wk + (size_t)DIM * DIM;
    const half_t* Wo  = Wh_out + (size_t)l * DIM * DIM;
    const half_t* Wfc = Wh_fc + (size_t)l * 4 * DIM * DIM;
    const half_t* Wpj = Wh_pj + (size_t)l * DIM * 4 * DIM;
    const float* bq  = in_proj_b + l * 3 * DIM;
    const float* bk  = bq + DIM;
    const float* bv_ = bq + 2 * DIM;
    const float* bo  = out_b + l * DIM;
    const float* bfc = fc_b + l * 4 * DIM;
    const float* bpj = proj_b + l * DIM;
    const float* l1w = ln1_w + l * DIM; const float* l1b = ln1_b + l * DIM;
    const float* l2w = ln2_w + l * DIM; const float* l2b = ln2_b + l * DIM;
    const float* l3w = ln3_w + l * DIM; const float* l3b = ln3_b + l * DIM;

    // kn = LN2(k) -> xA
    for (int r = wave; r < LQ; r += 8) ln_row_to_f16(krow(r), l2w, l2b, xA + (size_t)r * XS, lane);
    __syncthreads();
    { v8f acc[8] = {}; gemm_acc(xA, Wk, DIM, 0, DIM, acc, wave, lane); store_tiles_f16(Kf, acc, bk, false, wave, lane); }
    { v8f acc[8] = {}; gemm_acc(xA, Wv, DIM, 0, DIM, acc, wave, lane); store_tiles_f16(Vf, acc, bv_, false, wave, lane); }
    __syncthreads();
    // qn = LN1(q) -> xA
    for (int r = wave; r < LQ; r += 8) ln_row_to_f16(qld + (size_t)r * DIM, l1w, l1b, xA + (size_t)r * XS, lane);
    __syncthreads();
    { v8f acc[8] = {}; gemm_acc(xA, Wq, DIM, 0, DIM, acc, wave, lane); store_tiles_f16(Qf, acc, bq, false, wave, lane); }
    __syncthreads();

    // attention core (tiny: 17x17 per head) in f32 VALU
    for (int e = tid; e < NHEADS * LQ * LQ; e += 256) {
      int h = e / (LQ * LQ); int rem = e - h * LQ * LQ;
      int qi = rem / LQ; int ki = rem - qi * LQ;
      const half_t* qp = Qf + (size_t)qi * XS + h * HEADD;
      const half_t* kp = Kf + (size_t)ki * XS + h * HEADD;
      float s = 0.f;
#pragma unroll
      for (int d = 0; d < HEADD; ++d) s += (float)qp[d] * (float)kp[d];
      sc[e] = s * 0.125f;   // hd^-0.5, hd = 64
    }
    __syncthreads();
    if (tid < NHEADS * LQ) {
      float* row = sc + tid * LQ;
      float mx = row[0];
      for (int t = 1; t < LQ; ++t) mx = fmaxf(mx, row[t]);
      float sum = 0.f;
      for (int t = 0; t < LQ; ++t) { float e2 = expf(row[t] - mx); row[t] = e2; sum += e2; }
      float inv = 1.f / sum;
      for (int t = 0; t < LQ; ++t) row[t] *= inv;
    }
    __syncthreads();
    // context = probs @ V -> xA (A operand of out-proj)
    for (int e = tid; e < LQ * DIM; e += 256) {
      int qi = e >> 9; int col = e & (DIM - 1); int h = col >> 6;
      const float* prow = sc + (h * LQ + qi) * LQ;
      float s = 0.f;
#pragma unroll
      for (int ki = 0; ki < LQ; ++ki) s += prow[ki] * (float)Vf[(size_t)ki * XS + col];
      xA[(size_t)qi * XS + col] = (half_t)s;
    }
    __syncthreads();
    { v8f acc[8] = {}; gemm_acc(xA, Wo, DIM, 0, DIM, acc, wave, lane); add_tiles_to_q(qld, acc, bo, wave, lane); }
    __syncthreads();

    // MLP: h = silu(LN3(q) @ fc^T) in 4 chunks of 512; q += h @ proj^T
    for (int r = wave; r < LQ; r += 8) ln_row_to_f16(qld + (size_t)r * DIM, l3w, l3b, xA + (size_t)r * XS, lane);
    for (int e = tid; e < (32 - LQ) * XS; e += 256) xB[LQ * XS + e] = (half_t)0.f;  // xB pad rows
    __syncthreads();
    {
      v8f delta[8] = {};
      for (int c = 0; c < 4; ++c) {
        v8f hacc[8] = {};
        gemm_acc(xA, Wfc + (size_t)c * DIM * DIM, DIM, 0, DIM, hacc, wave, lane);
        store_tiles_f16(xB, hacc, bfc + c * DIM, true, wave, lane);
        __syncthreads();
        gemm_acc(xB, Wpj, 4 * DIM, c * DIM, DIM, delta, wave, lane);
        __syncthreads();
      }
      add_tiles_to_q(qld, delta, bpj, wave, lane);
    }
    __syncthreads();
  }

  // write (dir, i, j, 17, 512)
  float* op = out + ((size_t)dir * BN * BN + (size_t)pr) * (LQ * DIM);
  for (int e = tid; e < LQ * DIM; e += 256) op[e] = qld[e];
}

// ---------------------------------------------------------------------------
extern "C" void kernel_launch(void* const* d_in, const int* in_sizes, int n_in,
                              void* d_out, int out_size, void* d_ws, size_t ws_size,
                              hipStream_t stream) {
  const float* image_feature = (const float*)d_in[0];
  const float* image_tokens  = (const float*)d_in[1];
  const float* text_feature  = (const float*)d_in[2];
  const float* text_tokens   = (const float*)d_in[3];
  const int*   atte_mask     = (const int*)d_in[4];
  const float* img_cls       = (const float*)d_in[5];
  const float* txt_cls       = (const float*)d_in[6];
  const float* in_proj_w     = (const float*)d_in[7];
  const float* in_proj_b     = (const float*)d_in[8];
  const float* out_w         = (const float*)d_in[9];
  const float* out_b_        = (const float*)d_in[10];
  const float* ln1_w = (const float*)d_in[11];
  const float* ln1_b = (const float*)d_in[12];
  const float* ln2_w = (const float*)d_in[13];
  const float* ln2_b = (const float*)d_in[14];
  const float* ln3_w = (const float*)d_in[15];
  const float* ln3_b = (const float*)d_in[16];
  const float* fc_w  = (const float*)d_in[17];
  const float* fc_b  = (const float*)d_in[18];
  const float* proj_w = (const float*)d_in[19];
  const float* proj_b = (const float*)d_in[20];
  (void)in_sizes; (void)n_in; (void)out_size; (void)ws_size;

  // workspace carve (bytes)
  char* ws = (char*)d_ws;
  size_t oImgN = 0;                                     // 32*196*512 f32
  size_t oTxtN = oImgN + (size_t)IMG_ROWS * DIM * 4;    // 32*77*512 f32
  size_t oTM   = oTxtN + (size_t)TXT_ROWS * DIM * 4;    // 32*512 f32
  size_t oIS   = oTM   + (size_t)BN * DIM * 4;          // 32*512 f32
  size_t oImgS = oIS   + (size_t)BN * DIM * 4;          // 32*32*196 f32
  size_t oTxtS = oImgS + (size_t)BN * BN * PATCH * 4;   // 32*32*77 f32
  size_t oIdxI = oTxtS + (size_t)BN * BN * SEQ * 4;     // 32*32*16 i32
  size_t oIdxT = oIdxI + (size_t)BN * BN * TOPK * 4;
  size_t oWh   = oIdxT + (size_t)BN * BN * TOPK * 4;    // f16 weights (16B aligned)

  float* img_n   = (float*)(ws + oImgN);
  float* txt_n   = (float*)(ws + oTxtN);
  float* tmasked = (float*)(ws + oTM);
  float* img_sum = (float*)(ws + oIS);
  float* img_sc  = (float*)(ws + oImgS);
  float* txt_sc  = (float*)(ws + oTxtS);
  int*   idx_i   = (int*)(ws + oIdxI);
  int*   idx_t   = (int*)(ws + oIdxT);
  half_t* Wh     = (half_t*)(ws + oWh);

  const int n_in_proj = NLAYERS * 3 * DIM * DIM;   // 1,572,864
  const int n_out     = NLAYERS * DIM * DIM;       //   524,288
  const int n_fc      = NLAYERS * 4 * DIM * DIM;   // 2,097,152
  const int n_pj      = NLAYERS * DIM * 4 * DIM;   // 2,097,152
  half_t* Wh_in  = Wh;
  half_t* Wh_out = Wh_in + n_in_proj;
  half_t* Wh_fc  = Wh_out + n_out;
  half_t* Wh_pj  = Wh_fc + n_fc;

  // 1) pack weights f32 -> f16 (L2-resident working set, 12.6 MB)
  al_f32_to_f16<<<2048, 256, 0, stream>>>(in_proj_w, Wh_in, n_in_proj);
  al_f32_to_f16<<<2048, 256, 0, stream>>>(out_w,     Wh_out, n_out);
  al_f32_to_f16<<<2048, 256, 0, stream>>>(fc_w,      Wh_fc,  n_fc);
  al_f32_to_f16<<<2048, 256, 0, stream>>>(proj_w,    Wh_pj,  n_pj);

  // 2) L2-normalize tokens
  al_normalize<<<(IMG_ROWS + TXT_ROWS + 7) / 8, 256, 0, stream>>>(
      image_tokens, text_tokens, img_n, txt_n);

  // 3) masked text sum / image sum (collapses the (i,j,p,s) sim tensor)
  al_masksum<<<2 * BN, 256, 0, stream>>>(txt_n, img_n, atte_mask, tmasked, img_sum);

  // 4) score dot products
  const int nScoreWaves = BN * BN * PATCH + BN * BN * SEQ;  // 279,552
  al_scores<<<(nScoreWaves + 7) / 8, 256, 0, stream>>>(
      img_n, txt_n, tmasked, img_sum, img_sc, txt_sc);

  // 5) top-16 select + index sort
  al_topk<<<(2 * BN * BN + 255) / 256, 256, 0, stream>>>(img_sc, txt_sc, idx_i, idx_t);

  // 6) two-layer cross-attention for all 2048 (pair, direction) problems
  al_cross_attn<<<2 * BN * BN, 256, SMEM_BYTES, stream>>>(
      image_feature, text_feature, img_cls, txt_cls,
      in_proj_b, out_b_, ln1_w, ln1_b, ln2_w, ln2_b, ln3_w, ln3_b,
      fc_b, proj_b, img_n, txt_n, idx_i, idx_t,
      Wh_in, Wh_out, Wh_fc, Wh_pj, (float*)d_out);
}
